// StarGNNWithVirtualNode_48301202211283
// MI455X (gfx1250) — compile-verified, matrix operand order checked by reading, not run on hardware
//
#include <hip/hip_runtime.h>
#include <hip/hip_bf16.h>

typedef __attribute__((ext_vector_type(16))) _Float16 v16h;
typedef __attribute__((ext_vector_type(8)))  _Float16 v8h;
typedef __attribute__((ext_vector_type(8)))  float    v8f;
typedef __attribute__((ext_vector_type(4)))  float    v4f;

#define IN_F  256
#define HIDF  128   // H1*HID and layer-2 width
#define OUTF  128
#define COMBK 512   // IN_F + HIDF + OUTF

static __device__ __forceinline__ v8f wmma32(v16h a, v16h b, v8f c) {
  // D = A(16x32 f16) * B(32x16 f16) + C(16x16 f32)
  return __builtin_amdgcn_wmma_f32_16x16x32_f16(false, a, false, b, (short)0, c,
                                                false, false);
}

// Per-lane A fragment: rows striped per ISA layout; p -> arow + kb + khalf.
// elements 0..7 = p[0..7], 8..15 = p[16..23]; f32->f16 convert on load.
static __device__ __forceinline__ v16h load_a_frag(const float* __restrict__ p) {
  v4f x0 = *(const v4f*)(p);
  v4f x1 = *(const v4f*)(p + 4);
  v4f x2 = *(const v4f*)(p + 16);
  v4f x3 = *(const v4f*)(p + 20);
  v16h a;
#pragma unroll
  for (int i = 0; i < 4; ++i) {
    a[i]      = (_Float16)x0[i];
    a[4 + i]  = (_Float16)x1[i];
    a[8 + i]  = (_Float16)x2[i];
    a[12 + i] = (_Float16)x3[i];
  }
  return a;
}

// Per-lane B fragment from pre-transposed f16 weights: two 16B vector loads.
static __device__ __forceinline__ v16h load_b_frag(const _Float16* __restrict__ p) {
  v8h b0 = *(const v8h*)(p);
  v8h b1 = *(const v8h*)(p + 16);
  return __builtin_shufflevector(b0, b1, 0, 1, 2, 3, 4, 5, 6, 7,
                                 8, 9, 10, 11, 12, 13, 14, 15);
}

// ---------------- weight transpose + f32->f16 convert: Wt[n][k] = W[k][n] ----
__global__ void wconv_kernel(const float* __restrict__ W, _Float16* __restrict__ Wt,
                             int K, int Nout) {
  int idx = blockIdx.x * blockDim.x + threadIdx.x;
  if (idx >= K * Nout) return;
  int n = idx / K, k = idx % K;
  Wt[idx] = (_Float16)W[(size_t)k * Nout + n];
}

// ---------------- fills ------------------------------------------------------
__global__ void fill_f32_kernel(float* p, float v, size_t n) {
  size_t i = (size_t)blockIdx.x * blockDim.x + threadIdx.x;
  if (i < n) p[i] = v;
}
__global__ void fill_i32_kernel(int* p, int v, size_t n) {
  size_t i = (size_t)blockIdx.x * blockDim.x + threadIdx.x;
  if (i < n) p[i] = v;
}

// ---------------- WMMA GEMM: rows [0,M0) from A0, [M0,M) from A1 -------------
// A row-major f32 (ld = K), Bt = Nout x K row-major f16, D = M x Nout f32.
// OOB rows read a zero dummy row -> zero divergence in the hot loop.
// block: (32, Nout/16); one wave per 16x16 output tile; grid.x = ceil(M/16).
template <int K>
__global__ void gemm_rows2_kernel(const float* __restrict__ A0, int M0,
                                  const float* __restrict__ A1,
                                  int M,
                                  const _Float16* __restrict__ Bt,
                                  const float* __restrict__ bias,
                                  float* __restrict__ D, int Nout,
                                  const float* __restrict__ zrow) {
  const int lane  = threadIdx.x & 31;
  const int mbase = blockIdx.x * 16;
  const int row   = mbase + (lane & 15);
  const int khalf = (lane >> 4) * 8;
  const int ncol  = threadIdx.y * 16 + (lane & 15);

  const float* arow;
  if (row < M0)     arow = A0 + (size_t)row * K;
  else if (row < M) arow = A1 + (size_t)(row - M0) * K;
  else              arow = zrow;
  const _Float16* brow = Bt + (size_t)ncol * K + khalf;
  arow += khalf;

  v8f acc = {};
#pragma unroll
  for (int kb = 0; kb < K; kb += 32)
    acc = wmma32(load_a_frag(arow + kb), load_b_frag(brow + kb), acc);

  const int mlo = (lane >> 4) * 8;
#pragma unroll
  for (int v = 0; v < 8; ++v) {
    int m = mbase + mlo + v;
    if (m < M) {
      float r = acc[v];
      if (bias) r += bias[ncol];
      D[(size_t)m * Nout + ncol] = r;
    }
  }
}

// ---------------- WMMA GEMM over comb = [x | h1 | h2], K = 512 ---------------
__global__ void gemm_comb_kernel(const float* __restrict__ X,
                                 const float* __restrict__ H1a,
                                 const float* __restrict__ H2a,
                                 int M,
                                 const _Float16* __restrict__ Bt,  // 128 x 512
                                 const float* __restrict__ bias,
                                 float* __restrict__ D,
                                 const float* __restrict__ zrow) {
  const int lane  = threadIdx.x & 31;
  const int mbase = blockIdx.x * 16;
  const int row   = mbase + (lane & 15);
  const int khalf = (lane >> 4) * 8;
  const int ncol  = threadIdx.y * 16 + (lane & 15);
  const bool rok  = row < M;

  const float* xr  = (rok ? X   + (size_t)row * IN_F : zrow) + khalf;
  const float* h1r = (rok ? H1a + (size_t)row * HIDF : zrow) + khalf;
  const float* h2r = (rok ? H2a + (size_t)row * HIDF : zrow) + khalf;
  const _Float16* brow = Bt + (size_t)ncol * COMBK + khalf;

  v8f acc = {};
#pragma unroll
  for (int kb = 0; kb < COMBK; kb += 32) {
    const float* arow;
    int koff;
    if (kb < IN_F)             { arow = xr;  koff = kb; }
    else if (kb < IN_F + HIDF) { arow = h1r; koff = kb - IN_F; }
    else                       { arow = h2r; koff = kb - IN_F - HIDF; }
    acc = wmma32(load_a_frag(arow + koff), load_b_frag(brow + kb), acc);
  }

  const int mlo = (lane >> 4) * 8;
#pragma unroll
  for (int v = 0; v < 8; ++v) {
    int m = mbase + mlo + v;
    if (m < M) D[(size_t)m * OUTF + ncol] = acc[v] + bias[ncol];
  }
}

// ---------------- attention scores: es/ed[n,h] = <h_pre[n,h,:], a_*[h,:]> ----
template <int H, int C>
__global__ void att_scores_kernel(const float* __restrict__ Hpre,
                                  const float* __restrict__ a_src,
                                  const float* __restrict__ a_dst,
                                  float* __restrict__ es, float* __restrict__ ed,
                                  int Nn) {
  int idx = blockIdx.x * blockDim.x + threadIdx.x;
  if (idx >= Nn * H) return;
  int n = idx / H, h = idx % H;
  const float* hp = Hpre + (size_t)n * H * C + (size_t)h * C;
  float s = 0.f, d = 0.f;
#pragma unroll 8
  for (int c = 0; c < C; ++c) {
    float v = hp[c];
    s += v * a_src[h * C + c];
    d += v * a_dst[h * C + c];
  }
  es[idx] = s; ed[idx] = d;
}

// edge (s,d) synthesis: real edges then virtual edges (every node -> node N+v)
static __device__ __forceinline__ void edge_sd(int e, const int* __restrict__ esrc,
                                               const int* __restrict__ edst,
                                               int E, int N, int& s, int& d) {
  if (e < E) { s = esrc[e]; d = edst[e]; }
  else       { int q = e - E; s = q % N; d = N + q / N; }
}

static __device__ __forceinline__ int enc_f32(float x) {
  int i = __float_as_int(x);
  return i >= 0 ? i : (i ^ 0x7FFFFFFF);
}
static __device__ __forceinline__ float dec_f32(int i) {
  return __int_as_float(i >= 0 ? i : (i ^ 0x7FFFFFFF));
}

// pass 1: e = leaky_relu(es[src]+ed[dst]); stash logit; segment-max via atomicMax
__global__ void edge_max_kernel(const int* __restrict__ esrc, const int* __restrict__ edst,
                                int E, int N, int Etot,
                                const float* __restrict__ es, const float* __restrict__ ed,
                                float* __restrict__ eb, int* __restrict__ emax, int H) {
  int t = blockIdx.x * blockDim.x + threadIdx.x;
  if (t >= Etot * H) return;
  int e = t / H, h = t % H, s, d;
  edge_sd(e, esrc, edst, E, N, s, d);
  float v = es[s * H + h] + ed[d * H + h];
  v = v > 0.f ? v : 0.2f * v;
  eb[t] = v;
  atomicMax(&emax[d * H + h], enc_f32(v));
}

// pass 2: ee = exp(e - max[dst]); stash; segment-sum via float atomic add
__global__ void edge_sum_kernel(const int* __restrict__ esrc, const int* __restrict__ edst,
                                int E, int N, int Etot,
                                float* __restrict__ eb, const int* __restrict__ emax,
                                float* __restrict__ den, int H) {
  int t = blockIdx.x * blockDim.x + threadIdx.x;
  if (t >= Etot * H) return;
  int e = t / H, h = t % H, s, d;
  edge_sd(e, esrc, edst, E, N, s, d);
  float ee = __expf(eb[t] - dec_f32(emax[d * H + h]));
  eb[t] = ee;
  __hip_atomic_fetch_add(&den[d * H + h], ee, __ATOMIC_RELAXED, __HIP_MEMORY_SCOPE_AGENT);
}

// pass 3: acc[dst] += alpha * h_pre[src]; one wave per edge, 512B gather/edge
template <int H, int C>
__global__ void edge_scatter_kernel(const int* __restrict__ esrc, const int* __restrict__ edst,
                                    int E, int N, int Etot,
                                    const float* __restrict__ Hpre,
                                    const float* __restrict__ eb,
                                    const float* __restrict__ den,
                                    float* __restrict__ acc) {
  int e = blockIdx.x * (blockDim.x >> 5) + (threadIdx.x >> 5);
  if (e >= Etot) return;
  int lane = threadIdx.x & 31, s, d;
  edge_sd(e, esrc, edst, E, N, s, d);
  constexpr int F = H * C;  // 128
#pragma unroll
  for (int jj = 0; jj < F; jj += 32) {
    int j = jj + lane;
    int h = j / C;
    float alpha = eb[e * H + h] / den[d * H + h];
    float v = Hpre[(size_t)s * F + j] * alpha;
    __hip_atomic_fetch_add(&acc[(size_t)d * F + j], v,
                           __ATOMIC_RELAXED, __HIP_MEMORY_SCOPE_AGENT);
  }
}

// bias (+ optional ELU), in place
__global__ void finalize_kernel(float* __restrict__ a, const float* __restrict__ bias,
                                size_t n, int F, int do_elu) {
  size_t i = (size_t)blockIdx.x * blockDim.x + threadIdx.x;
  if (i >= n) return;
  float v = a[i] + bias[i % F];
  if (do_elu) v = v > 0.f ? v : expm1f(v);
  a[i] = v;
}

extern "C" void kernel_launch(void* const* d_in, const int* in_sizes, int n_in,
                              void* d_out, int out_size, void* d_ws, size_t ws_size,
                              hipStream_t stream) {
  (void)n_in; (void)out_size; (void)ws_size;
  const float* x    = (const float*)d_in[0];
  const int*   ei   = (const int*)d_in[1];
  const float* vemb = (const float*)d_in[3];
  const float* W1   = (const float*)d_in[4];
  const float* a1s  = (const float*)d_in[5];
  const float* a1d  = (const float*)d_in[6];
  const float* b1   = (const float*)d_in[7];
  const float* W2   = (const float*)d_in[8];
  const float* a2s  = (const float*)d_in[9];
  const float* a2d  = (const float*)d_in[10];
  const float* b2   = (const float*)d_in[11];
  const float* Wj   = (const float*)d_in[12];
  const float* bj   = (const float*)d_in[13];

  const int N    = in_sizes[0] / IN_F;     // 50000
  const int E    = in_sizes[1] / 2;        // 800000
  const int V    = in_sizes[3] / IN_F;     // 1
  const int NA   = N + V;                  // augmented node count
  const int Etot = E + N * V;              // real + virtual edges
  const int* esrc = ei;
  const int* edst = ei + E;

  // ---- carve workspace (256B aligned slabs) ----
  char* wptr = (char*)d_ws;
  auto carve = [&](size_t bytes) -> void* {
    void* p = (void*)wptr;
    wptr += (bytes + 255) & ~(size_t)255;
    return p;
  };
  float*    h1p  = (float*)carve((size_t)NA * HIDF * 4);
  float*    h1   = (float*)carve((size_t)NA * HIDF * 4);  // acc1, finalized in place
  float*    h2p  = (float*)carve((size_t)NA * HIDF * 4);
  float*    h2   = (float*)carve((size_t)NA * HIDF * 4);  // acc2, finalized in place
  _Float16* W1t  = (_Float16*)carve((size_t)IN_F * HIDF * 2);
  _Float16* W2t  = (_Float16*)carve((size_t)HIDF * HIDF * 2);
  _Float16* Wjt  = (_Float16*)carve((size_t)COMBK * OUTF * 2);
  float*    zrow = (float*)carve((size_t)COMBK * 4);      // zero dummy row
  float*    es1  = (float*)carve((size_t)NA * 2 * 4);
  float*    ed1  = (float*)carve((size_t)NA * 2 * 4);
  int*      emx1 = (int*)carve((size_t)NA * 2 * 4);
  float*    den1 = (float*)carve((size_t)NA * 2 * 4);
  float*    eb1  = (float*)carve((size_t)Etot * 2 * 4);
  float*    es2  = (float*)carve((size_t)NA * 4);
  float*    ed2  = (float*)carve((size_t)NA * 4);
  int*      emx2 = (int*)carve((size_t)NA * 4);
  float*    den2 = (float*)carve((size_t)NA * 4);
  float*    eb2  = (float*)carve((size_t)Etot * 4);

  const int TB = 256;
  auto blks = [](size_t n, int tb) { return (unsigned)((n + tb - 1) / tb); };
  const dim3 gblock(32, HIDF / 16);            // 8 waves/block, one n-tile each
  const unsigned gtilesA = (unsigned)((NA + 15) / 16);
  const unsigned gtilesN = (unsigned)((N + 15) / 16);
  const int NEG_INF_ENC = (int)0x80000000;

  // weights -> transposed f16; zero dummy row
  wconv_kernel<<<blks((size_t)IN_F * HIDF, TB), TB, 0, stream>>>(W1, W1t, IN_F, HIDF);
  wconv_kernel<<<blks((size_t)HIDF * HIDF, TB), TB, 0, stream>>>(W2, W2t, HIDF, HIDF);
  wconv_kernel<<<blks((size_t)COMBK * OUTF, TB), TB, 0, stream>>>(Wj, Wjt, COMBK, OUTF);
  fill_f32_kernel<<<blks((size_t)COMBK, TB), TB, 0, stream>>>(zrow, 0.f, (size_t)COMBK);

  // ================= layer 1 (H=2, C=64) =================
  gemm_rows2_kernel<IN_F><<<gtilesA, gblock, 0, stream>>>(x, N, vemb, NA, W1t,
                                                          nullptr, h1p, HIDF, zrow);
  att_scores_kernel<2, 64><<<blks((size_t)NA * 2, TB), TB, 0, stream>>>(h1p, a1s, a1d,
                                                                        es1, ed1, NA);
  fill_i32_kernel<<<blks((size_t)NA * 2, TB), TB, 0, stream>>>(emx1, NEG_INF_ENC, (size_t)NA * 2);
  fill_f32_kernel<<<blks((size_t)NA * 2, TB), TB, 0, stream>>>(den1, 0.f, (size_t)NA * 2);
  fill_f32_kernel<<<blks((size_t)NA * HIDF, TB), TB, 0, stream>>>(h1, 0.f, (size_t)NA * HIDF);
  edge_max_kernel<<<blks((size_t)Etot * 2, TB), TB, 0, stream>>>(esrc, edst, E, N, Etot,
                                                                 es1, ed1, eb1, emx1, 2);
  edge_sum_kernel<<<blks((size_t)Etot * 2, TB), TB, 0, stream>>>(esrc, edst, E, N, Etot,
                                                                 eb1, emx1, den1, 2);
  edge_scatter_kernel<2, 64><<<blks((size_t)Etot * 32, TB), TB, 0, stream>>>(esrc, edst, E, N,
                                                                             Etot, h1p, eb1,
                                                                             den1, h1);
  finalize_kernel<<<blks((size_t)NA * HIDF, TB), TB, 0, stream>>>(h1, b1, (size_t)NA * HIDF,
                                                                  HIDF, 1);

  // ================= layer 2 (H=1, C=128) =================
  gemm_rows2_kernel<HIDF><<<gtilesA, gblock, 0, stream>>>(h1, NA, nullptr, NA, W2t,
                                                          nullptr, h2p, HIDF, zrow);
  att_scores_kernel<1, 128><<<blks((size_t)NA, TB), TB, 0, stream>>>(h2p, a2s, a2d,
                                                                     es2, ed2, NA);
  fill_i32_kernel<<<blks((size_t)NA, TB), TB, 0, stream>>>(emx2, NEG_INF_ENC, (size_t)NA);
  fill_f32_kernel<<<blks((size_t)NA, TB), TB, 0, stream>>>(den2, 0.f, (size_t)NA);
  fill_f32_kernel<<<blks((size_t)NA * HIDF, TB), TB, 0, stream>>>(h2, 0.f, (size_t)NA * HIDF);
  edge_max_kernel<<<blks((size_t)Etot, TB), TB, 0, stream>>>(esrc, edst, E, N, Etot,
                                                             es2, ed2, eb2, emx2, 1);
  edge_sum_kernel<<<blks((size_t)Etot, TB), TB, 0, stream>>>(esrc, edst, E, N, Etot,
                                                             eb2, emx2, den2, 1);
  edge_scatter_kernel<1, 128><<<blks((size_t)Etot * 32, TB), TB, 0, stream>>>(esrc, edst, E, N,
                                                                              Etot, h2p, eb2,
                                                                              den2, h2);
  finalize_kernel<<<blks((size_t)NA * HIDF, TB), TB, 0, stream>>>(h2, b2, (size_t)NA * HIDF,
                                                                  HIDF, 0);

  // ================= joint projection =================
  gemm_comb_kernel<<<gtilesN, gblock, 0, stream>>>(x, h1, h2, N, Wjt, bj,
                                                   (float*)d_out, zrow);
}
